// MultiplePositivesLoss_11218454577736
// MI455X (gfx1250) — matrix operand order repaired and checked by reading, not compiled, not run on hardware
//
#include <hip/hip_runtime.h>
#include <hip/hip_bf16.h>

// ---------------------------------------------------------------------------
// MultiplePositivesLoss, fused flash-style on gfx1250 (wave32, WMMA).
//   sims = Q @ A^T  (8192x8192x128) via split-bf16 WMMA (3x v_wmma_f32_16x16x32_bf16)
//   online per-row: max / sum(exp) / sum(w*exp) / argmax  -> never materialize sims
//   pass 2: merge column-split partials, per-row loss/correct, reduce to 2 scalars
// ---------------------------------------------------------------------------

#define B_N        8192
#define D_K        128
#define M_TILE     64            // rows per workgroup (16 per wave, 4 waves)
#define N_TILE     64            // columns per chunk
#define BLOCK      128
#define COL_SPLIT  4
#define COLS_PS    (B_N / COL_SPLIT)
#define CHUNKS     (COLS_PS / N_TILE)
#define LDSTR      136           // LDS row stride in bf16 elems (pad vs 64 banks)
#define RANK_W     0.1f
#define LEPS       1e-8f

typedef __attribute__((ext_vector_type(16))) __bf16 v16bf;
typedef __attribute__((ext_vector_type(8)))  float  v8f;

union FragBF { v16bf v; unsigned int u[8]; };
union AccF   { v8f   v; float        f[8]; };

static __device__ __forceinline__ unsigned short f2bf(float f) {
  unsigned u = __builtin_bit_cast(unsigned, f);
  u += 0x7FFFu + ((u >> 16) & 1u);            // round-to-nearest-even
  return (unsigned short)(u >> 16);
}
static __device__ __forceinline__ float bf2f(unsigned short h) {
  unsigned u = ((unsigned)h) << 16;
  return __builtin_bit_cast(float, u);
}

__global__ __launch_bounds__(BLOCK) void mpl_gemm_softmax(
    const float* __restrict__ Q, const float* __restrict__ A,
    const int* __restrict__ qid, const float* __restrict__ ranks,
    float* __restrict__ pm, float* __restrict__ pden, float* __restrict__ pnum,
    float* __restrict__ pbv, int* __restrict__ pbj)
{
  __shared__ unsigned short qhi[M_TILE * LDSTR];
  __shared__ unsigned short qlo[M_TILE * LDSTR];
  __shared__ unsigned short ahi[N_TILE * LDSTR];
  __shared__ unsigned short alo[N_TILE * LDSTR];
  __shared__ int   s_qid_row[M_TILE];
  __shared__ int   s_qid_col[N_TILE];
  __shared__ float s_rank_col[N_TILE];

  const int tid  = threadIdx.x;
  const int lane = tid & 31;
  const int wave = tid >> 5;
  const int grp  = lane & 15;     // position inside 16-lane group
  const int half = lane >> 4;     // which half of the wave
  const int row_base = blockIdx.x * M_TILE;
  const int col_base = blockIdx.y * COLS_PS;
  const int split    = blockIdx.y;

  // ---- stage Q rows once: hi/lo bf16 split into LDS --------------------
  for (int i = tid; i < M_TILE * D_K; i += BLOCK) {
    int r = i >> 7, k = i & (D_K - 1);
    float f = Q[(size_t)(row_base + r) * D_K + k];
    unsigned short h = f2bf(f);
    qhi[r * LDSTR + k] = h;
    qlo[r * LDSTR + k] = f2bf(f - bf2f(h));
  }
  if (tid < M_TILE) s_qid_row[tid] = qid[row_base + tid];
  __syncthreads();

  int rowqid[8];
#pragma unroll
  for (int r = 0; r < 8; ++r) rowqid[r] = s_qid_row[wave * 16 + r + 8 * half];

  // per-lane online-softmax state for the 8 rows this (lane,half) touches
  float m_run[8], den[8], num[8], bestv[8]; int bestj[8];
#pragma unroll
  for (int r = 0; r < 8; ++r) {
    m_run[r] = -3.0e38f; den[r] = 0.f; num[r] = 0.f;
    bestv[r] = -3.0e38f; bestj[r] = 0x7fffffff;
  }

  for (int ch = 0; ch < CHUNKS; ++ch) {
    const int c0 = col_base + ch * N_TILE;

    __syncthreads();
    for (int i = tid; i < N_TILE * D_K; i += BLOCK) {
      int r = i >> 7, k = i & (D_K - 1);
      float f = A[(size_t)(c0 + r) * D_K + k];
      unsigned short h = f2bf(f);
      ahi[r * LDSTR + k] = h;
      alo[r * LDSTR + k] = f2bf(f - bf2f(h));
    }
    if (tid < N_TILE) { s_qid_col[tid] = qid[c0 + tid]; s_rank_col[tid] = ranks[c0 + tid]; }
    __syncthreads();

    // ---- 16x64 sims tile for this wave, split-bf16 f32 accumulate ------
    v8f acc[4];
#pragma unroll
    for (int s = 0; s < 4; ++s) { AccF z; 
#pragma unroll
      for (int e = 0; e < 8; ++e) z.f[e] = 0.f; acc[s] = z.v; }

#pragma unroll
    for (int kc = 0; kc < 4; ++kc) {
      const int kb = kc * 32;
      // A-operand (Q rows), ISA 16-bit A 16x32 layout:
      //  half0 lane: K = {0..7, 16..23}; half1 lane: K = {8..15, 24..31}
      FragBF qh, ql;
#pragma unroll
      for (int v = 0; v < 8; ++v) {
        int k0 = kb + ((v < 4) ? 2 * v : 16 + 2 * (v - 4)) + (half ? 8 : 0);
        qh.u[v] = *(const unsigned int*)&qhi[(wave * 16 + grp) * LDSTR + k0];
        ql.u[v] = *(const unsigned int*)&qlo[(wave * 16 + grp) * LDSTR + k0];
      }
#pragma unroll
      for (int s = 0; s < 4; ++s) {
        // B-operand (A-embedding cols), ISA 16-bit B 32x16 layout:
        //  half0 lane: K = {0..7, 16..23}; half1 lane: K = {8..15, 24..31}, N = lane&15
        FragBF bh, bl;
#pragma unroll
        for (int v = 0; v < 8; ++v) {
          int k0 = kb + ((v < 4) ? 2 * v : 16 + 2 * (v - 4)) + (half ? 8 : 0);
          bh.u[v] = *(const unsigned int*)&ahi[(s * 16 + grp) * LDSTR + k0];
          bl.u[v] = *(const unsigned int*)&alo[(s * 16 + grp) * LDSTR + k0];
        }
        acc[s] = __builtin_amdgcn_wmma_f32_16x16x32_bf16(false, qh.v, false, bh.v,
                                                         (short)0, acc[s], false, false);
        acc[s] = __builtin_amdgcn_wmma_f32_16x16x32_bf16(false, qh.v, false, bl.v,
                                                         (short)0, acc[s], false, false);
        acc[s] = __builtin_amdgcn_wmma_f32_16x16x32_bf16(false, ql.v, false, bh.v,
                                                         (short)0, acc[s], false, false);
      }
    }

    // ---- online softmax / weighted-num / argmax update ------------------
    AccF ua[4];
#pragma unroll
    for (int s = 0; s < 4; ++s) ua[s].v = acc[s];

#pragma unroll
    for (int r = 0; r < 8; ++r) {
      float x0 = ua[0].f[r], x1 = ua[1].f[r], x2 = ua[2].f[r], x3 = ua[3].f[r];
      float cm = fmaxf(fmaxf(x0, x1), fmaxf(x2, x3));
#pragma unroll
      for (int off = 1; off <= 8; off <<= 1)
        cm = fmaxf(cm, __shfl_xor(cm, off, 32));
      float mnew = fmaxf(m_run[r], cm);

      float e0 = __expf(x0 - mnew), e1 = __expf(x1 - mnew);
      float e2 = __expf(x2 - mnew), e3 = __expf(x3 - mnew);
      float dsum = e0 + e1 + e2 + e3;

      int myq = rowqid[r];
      float w0 = (s_qid_col[grp]      == myq) ? (1.f - RANK_W * s_rank_col[grp])      : 0.f;
      float w1 = (s_qid_col[16 + grp] == myq) ? (1.f - RANK_W * s_rank_col[16 + grp]) : 0.f;
      float w2 = (s_qid_col[32 + grp] == myq) ? (1.f - RANK_W * s_rank_col[32 + grp]) : 0.f;
      float w3 = (s_qid_col[48 + grp] == myq) ? (1.f - RANK_W * s_rank_col[48 + grp]) : 0.f;
      float nsum = w0 * e0 + w1 * e1 + w2 * e2 + w3 * e3;

      float bv = x0; int bj = c0 + grp;                 // first-index tie-break
      if (x1 > bv) { bv = x1; bj = c0 + 16 + grp; }
      if (x2 > bv) { bv = x2; bj = c0 + 32 + grp; }
      if (x3 > bv) { bv = x3; bj = c0 + 48 + grp; }

#pragma unroll
      for (int off = 1; off <= 8; off <<= 1) {
        dsum += __shfl_xor(dsum, off, 32);
        nsum += __shfl_xor(nsum, off, 32);
        float ov = __shfl_xor(bv, off, 32);
        int   oj = __shfl_xor(bj, off, 32);
        if (ov > bv || (ov == bv && oj < bj)) { bv = ov; bj = oj; }
      }

      float sc = __expf(m_run[r] - mnew);
      den[r] = den[r] * sc + dsum;
      num[r] = num[r] * sc + nsum;
      m_run[r] = mnew;
      if (bv > bestv[r] || (bv == bestv[r] && bj < bestj[r])) { bestv[r] = bv; bestj[r] = bj; }
    }
  }

  // ---- write per-(row,split) partial states ------------------------------
  if (grp == 0) {
#pragma unroll
    for (int r = 0; r < 8; ++r) {
      int row = row_base + wave * 16 + r + 8 * half;
      int idx = row * COL_SPLIT + split;
      pm[idx]  = m_run[r]; pden[idx] = den[r]; pnum[idx] = num[r];
      pbv[idx] = bestv[r]; pbj[idx]  = bestj[r];
    }
  }
}

__global__ __launch_bounds__(1024) void mpl_finalize(
    const int* __restrict__ qid, const float* __restrict__ ranks,
    const float* __restrict__ pm, const float* __restrict__ pden,
    const float* __restrict__ pnum, const float* __restrict__ pbv,
    const int* __restrict__ pbj, float* __restrict__ out)
{
  __shared__ float sl[1024];
  __shared__ float sc[1024];
  const int tid = threadIdx.x;
  float lsum = 0.f, csum = 0.f;

  for (int row = tid; row < B_N; row += 1024) {
    float m = -3.0e38f;
#pragma unroll
    for (int s = 0; s < COL_SPLIT; ++s) m = fmaxf(m, pm[row * COL_SPLIT + s]);
    float den = 0.f, num = 0.f, bv = -3.0e38f; int bj = 0x7fffffff;
#pragma unroll
    for (int s = 0; s < COL_SPLIT; ++s) {       // ascending split => ascending j
      int idx = row * COL_SPLIT + s;
      float scale = __expf(pm[idx] - m);
      den += pden[idx] * scale;
      num += pnum[idx] * scale;
      float v = pbv[idx]; int j = pbj[idx];
      if (v > bv || (v == bv && j < bj)) { bv = v; bj = j; }
    }
    lsum += -logf(num / den + LEPS);
    float w = (qid[bj] == qid[row]) ? (1.f - RANK_W * ranks[bj]) : 0.f;
    csum += (w > 0.f) ? 1.f : 0.f;
  }

  sl[tid] = lsum; sc[tid] = csum;
  __syncthreads();
  for (int off = 512; off > 0; off >>= 1) {
    if (tid < off) { sl[tid] += sl[tid + off]; sc[tid] += sc[tid + off]; }
    __syncthreads();
  }
  if (tid == 0) {
    out[0] = sl[0] / (float)B_N;   // loss
    out[1] = sc[0] / (float)B_N;   // accuracy
  }
}

extern "C" void kernel_launch(void* const* d_in, const int* in_sizes, int n_in,
                              void* d_out, int out_size, void* d_ws, size_t ws_size,
                              hipStream_t stream) {
  const float* q     = (const float*)d_in[0];
  const float* a     = (const float*)d_in[1];
  const int*   qids  = (const int*)d_in[2];
  const float* ranks = (const float*)d_in[3];
  float* out = (float*)d_out;

  // workspace: 5 arrays of [B_N * COL_SPLIT] partials (~640 KB)
  float* pm   = (float*)d_ws;
  float* pden = pm   + B_N * COL_SPLIT;
  float* pnum = pden + B_N * COL_SPLIT;
  float* pbv  = pnum + B_N * COL_SPLIT;
  int*   pbj  = (int*)(pbv + B_N * COL_SPLIT);

  dim3 grid(B_N / M_TILE, COL_SPLIT);           // 128 x 4 = 512 workgroups
  mpl_gemm_softmax<<<grid, BLOCK, 0, stream>>>(q, a, qids, ranks,
                                               pm, pden, pnum, pbv, pbj);
  mpl_finalize<<<1, 1024, 0, stream>>>(qids, ranks, pm, pden, pnum, pbv, pbj, out);
}
